// Quantize_90074054132072
// MI455X (gfx1250) — matrix operand order repaired and checked by reading, not compiled, not run on hardware
//
#include <hip/hip_runtime.h>

typedef __attribute__((ext_vector_type(16))) __bf16 v16bf;
typedef __attribute__((ext_vector_type(8)))  __bf16 v8bf;
typedef __attribute__((ext_vector_type(8)))  float  v8f;

#define N_ 16384
#define D_ 256
#define K_ 4096

#define LOG2E 1.4426950408889634f
#define LN2   0.6931471805599453f
#define EPSQ  1e-20f

__device__ __forceinline__ unsigned short f2bf(float f) {
  unsigned int v = __float_as_uint(f);
  v += 0x7FFFu + ((v >> 16) & 1u);   // round-to-nearest-even
  return (unsigned short)(v >> 16);
}
__device__ __forceinline__ float bf2f(unsigned short b) {
  return __uint_as_float(((unsigned int)b) << 16);
}

union Frag16 {
  unsigned short s[16];
  v16bf v;
  struct { v8bf lo, hi; } h;
};

union V4U {
  uint4 q;
  unsigned short us[8];
};

// ---------------------------------------------------------------------------
// Prep: codebook f32 [K][D] -> bf16 cb [K][D], bf16 cbT [D][K], c_sq [K]
// ---------------------------------------------------------------------------
__global__ __launch_bounds__(256) void vq_prep(const float* __restrict__ codebook,
                                               unsigned short* __restrict__ cb,
                                               unsigned short* __restrict__ cbT,
                                               float* __restrict__ csq) {
  const int k = blockIdx.x;
  const int d = threadIdx.x;
  float v = codebook[k * D_ + d];
  unsigned short b = f2bf(v);
  cb[k * D_ + d] = b;
  cbT[d * K_ + k] = b;
  __shared__ float red[256];
  red[d] = v * v;
  __syncthreads();
  for (int s = 128; s > 0; s >>= 1) {
    if (d < s) red[d] += red[d + s];
    __syncthreads();
  }
  if (d == 0) csq[k] = red[0];
}

// ---------------------------------------------------------------------------
// Fused: logits GEMM -> softmax/argmax in LDS -> emb GEMM. 16 rows per block.
// Logits kept in log2 domain: l2 = (s*log2e - log2(-log2(u+e)*ln2 + e)) * invT
// so softmax weight = exp2(l2 - rowmax(l2)).
// ---------------------------------------------------------------------------
__global__ __launch_bounds__(256) void vq_main(const float* __restrict__ x,
                                               const float* __restrict__ u,
                                               const float* __restrict__ temp,
                                               const unsigned short* __restrict__ cb_,
                                               const unsigned short* __restrict__ cbT_,
                                               const float* __restrict__ csq,
                                               float* __restrict__ emb,
                                               float* __restrict__ idsp) {
  const __bf16* cb  = (const __bf16*)cb_;
  const __bf16* cbT = (const __bf16*)cbT_;

  __shared__ unsigned short Wlds[16 * K_];   // 128 KB: log2-logits -> exp2 weights
  __shared__ float args_s[16 * 128];         // argmax candidates (score)
  __shared__ int   args_k[16 * 128];         // argmax candidates (index)
  __shared__ float args_m[16 * 128];         // per-lane row max of l2
  __shared__ float red[16 * 16];             // row reductions
  __shared__ float rowstat[32];              // [0..15]=rowmax, [16..31]=1/sum

  const int tid  = threadIdx.x;
  const int wave = tid >> 5;
  const int lane = tid & 31;
  const int hi   = lane >> 4;    // half-wave selector
  const int ln   = lane & 15;
  const int row0 = blockIdx.x * 16;
  const float invT = 1.0f / temp[0];
  const float c1   = LOG2E * invT;           // s * c1
  const float c2   = invT;                   // -L2 * c2

  // ---- Phase 0: build bf16 A fragments of the 16x256 x-tile (held in regs) ----
  Frag16 afrag[8];
  const float* xr = x + (size_t)(row0 + ln) * D_;
  #pragma unroll
  for (int dc = 0; dc < 8; ++dc) {
    const float* p = xr + dc * 32 + 16 * hi;
    float4 a0 = *(const float4*)(p);
    float4 a1 = *(const float4*)(p + 4);
    float4 b0 = *(const float4*)(p + 8);
    float4 b1 = *(const float4*)(p + 12);
    afrag[dc].s[0]  = f2bf(a0.x); afrag[dc].s[1]  = f2bf(a0.y);
    afrag[dc].s[2]  = f2bf(a0.z); afrag[dc].s[3]  = f2bf(a0.w);
    afrag[dc].s[4]  = f2bf(a1.x); afrag[dc].s[5]  = f2bf(a1.y);
    afrag[dc].s[6]  = f2bf(a1.z); afrag[dc].s[7]  = f2bf(a1.w);
    afrag[dc].s[8]  = f2bf(b0.x); afrag[dc].s[9]  = f2bf(b0.y);
    afrag[dc].s[10] = f2bf(b0.z); afrag[dc].s[11] = f2bf(b0.w);
    afrag[dc].s[12] = f2bf(b1.x); afrag[dc].s[13] = f2bf(b1.y);
    afrag[dc].s[14] = f2bf(b1.z); afrag[dc].s[15] = f2bf(b1.w);
  }

  // ---- Phase 1: s = 2*x.c - c^2 (x^2 cancels); l2 = (s*log2e - L2) * invT ----
  float bestS[8];
  int   bestK[8];
  float mxL[8];
  #pragma unroll
  for (int r = 0; r < 8; ++r) { bestS[r] = -3.4e38f; bestK[r] = 0; mxL[r] = -3.4e38f; }

  for (int t = 0; t < 32; ++t) {
    const int k0 = (wave * 32 + t) * 16;
    const int kk = k0 + ln;
    v8f acc0 = {};
    v8f acc1 = {};
    const __bf16* bbase = cb + (size_t)kk * D_ + 16 * hi;
    #pragma unroll
    for (int dc = 0; dc < 8; dc += 2) {
      v16bf bv0 = *(const v16bf*)(bbase + dc * 32);
      v16bf bv1 = *(const v16bf*)(bbase + dc * 32 + 32);
      acc0 = __builtin_amdgcn_wmma_f32_16x16x32_bf16(
          false, afrag[dc].v, false, bv0, (short)0, acc0, false, false);
      acc1 = __builtin_amdgcn_wmma_f32_16x16x32_bf16(
          false, afrag[dc + 1].v, false, bv1, (short)0, acc1, false, false);
    }
    const float cs = csq[kk];
    const float* ubase = u + (size_t)(row0 + 8 * hi) * K_ + kk;
    #pragma unroll
    for (int r = 0; r < 8; ++r) {
      float s = fmaf(2.0f, acc0[r] + acc1[r], -cs);   // = -dist + x^2 (const/row)
      if (s > bestS[r]) { bestS[r] = s; bestK[r] = kk; }
      float uu = ubase[(size_t)r * K_];
      float L1 = __log2f(uu + EPSQ);                  // log2(u+e)
      float tt = fmaf(L1, -LN2, EPSQ);                // -ln(u+e) + e
      float L2 = __log2f(tt);                         // log2(.)  (gumbel/ln2 = -L2)
      float l2 = fmaf(s, c1, -L2 * c2);
      if (l2 > mxL[r]) mxL[r] = l2;
      Wlds[(r + 8 * hi) * K_ + kk] = f2bf(l2);
    }
  }
  #pragma unroll
  for (int r = 0; r < 8; ++r) {
    const int p = (r + 8 * hi) * 128 + wave * 16 + ln;
    args_s[p] = bestS[r];
    args_k[p] = bestK[r];
    args_m[p] = mxL[r];
  }
  __syncthreads();

  // ---- Phase 2: finalize argmax + rowmax, then one vectorized exp2 pass ----
  const int row  = tid >> 4;   // 16 threads cooperate per row
  const int slot = tid & 15;

  float bs = -3.4e38f; int bk = 0; float mx = -3.4e38f;
  #pragma unroll
  for (int j = 0; j < 8; ++j) {
    const int idx = row * 128 + slot * 8 + j;
    float s = args_s[idx];
    if (s > bs) { bs = s; bk = args_k[idx]; }
    mx = fmaxf(mx, args_m[idx]);
  }
  __syncthreads();
  args_s[row * 128 + slot] = bs;
  args_k[row * 128 + slot] = bk;
  red[row * 16 + slot] = mx;
  __syncthreads();
  if (slot == 0) {
    float b2 = -3.4e38f; int k2 = 0; float m2 = -3.4e38f;
    #pragma unroll
    for (int j = 0; j < 16; ++j) {
      float s = args_s[row * 128 + j];
      if (s > b2) { b2 = s; k2 = args_k[row * 128 + j]; }
      m2 = fmaxf(m2, red[row * 16 + j]);
    }
    idsp[row0 + row] = (float)k2;
    rowstat[row] = m2;
  }
  __syncthreads();

  const float rm = rowstat[row];
  unsigned short* wrow = &Wlds[row * K_ + slot * 256];
  float sum = 0.0f;
  for (int j = 0; j < 32; ++j) {
    const int jj = (slot + j) & 31;            // rotate to spread LDS banks
    V4U v;
    v.q = *(const uint4*)(wrow + jj * 8);
    #pragma unroll
    for (int e = 0; e < 8; ++e) {
      float w = __builtin_amdgcn_exp2f(bf2f(v.us[e]) - rm);  // unnormalized weight
      sum += w;
      v.us[e] = f2bf(w);
    }
    *(uint4*)(wrow + jj * 8) = v.q;
  }
  red[row * 16 + slot] = sum;
  __syncthreads();
  if (slot == 0) {
    float s = 0.0f;
    #pragma unroll
    for (int j = 0; j < 16; ++j) s += red[row * 16 + j];
    rowstat[16 + row] = 1.0f / s;
  }
  __syncthreads();

  // ---- Phase 3: emb = softmax(W) @ codebook, A from LDS, 1/sum folded in ----
  #pragma unroll
  for (int nt = 0; nt < 2; ++nt) {
    const int d0 = (wave * 2 + nt) * 16;
    v8f acc0 = {};
    v8f acc1 = {};
    const unsigned short* abase = &Wlds[ln * K_ + 16 * hi];
    const __bf16* bbase = cbT + (size_t)(d0 + ln) * K_ + 16 * hi;
    #pragma unroll 2
    for (int kc = 0; kc < 128; kc += 2) {
      Frag16 af0, af1;
      af0.h.lo = *(const v8bf*)(abase + kc * 32);
      af0.h.hi = *(const v8bf*)(abase + kc * 32 + 8);
      af1.h.lo = *(const v8bf*)(abase + kc * 32 + 32);
      af1.h.hi = *(const v8bf*)(abase + kc * 32 + 40);
      v16bf bv0 = *(const v16bf*)(bbase + kc * 32);
      v16bf bv1 = *(const v16bf*)(bbase + kc * 32 + 32);
      acc0 = __builtin_amdgcn_wmma_f32_16x16x32_bf16(
          false, af0.v, false, bv0, (short)0, acc0, false, false);
      acc1 = __builtin_amdgcn_wmma_f32_16x16x32_bf16(
          false, af1.v, false, bv1, (short)0, acc1, false, false);
    }
    #pragma unroll
    for (int r = 0; r < 8; ++r) {
      const int m = r + 8 * hi;
      emb[(size_t)(row0 + m) * D_ + d0 + ln] = (acc0[r] + acc1[r]) * rowstat[16 + m];
    }
  }
}

// ---------------------------------------------------------------------------
extern "C" void kernel_launch(void* const* d_in, const int* in_sizes, int n_in,
                              void* d_out, int out_size, void* d_ws, size_t ws_size,
                              hipStream_t stream) {
  const float* x        = (const float*)d_in[0];   // [N, D]
  const float* codebook = (const float*)d_in[1];   // [K, D]
  const float* u        = (const float*)d_in[2];   // [N, K]
  const float* temp     = (const float*)d_in[3];   // scalar

  float* out  = (float*)d_out;
  float* embp = out;                         // [N, D]
  float* idsp = out + (size_t)N_ * D_;       // [N]

  char* ws = (char*)d_ws;
  unsigned short* cb  = (unsigned short*)(ws);                          // 2 MB
  unsigned short* cbT = (unsigned short*)(ws + (size_t)K_ * D_ * 2);    // 2 MB
  float*          csq = (float*)(ws + (size_t)K_ * D_ * 4);             // 16 KB

  vq_prep<<<K_, 256, 0, stream>>>(codebook, cb, cbT, csq);
  vq_main<<<N_ / 16, 256, 0, stream>>>(x, u, temp, cb, cbT, csq, embp, idsp);
}
